// TensorFieldNetwork_34677565948872
// MI455X (gfx1250) — matrix operand order repaired and checked by reading, not compiled, not run on hardware
//
#include <hip/hip_runtime.h>
#include <hip/hip_bf16.h>

typedef __attribute__((ext_vector_type(16))) _Float16 v16h;
typedef __attribute__((ext_vector_type(8)))  float    v8f;

#define NBATCH 2
#define N 128
#define RBF 16
#define HID 32
#define H0 64
#define H1 16
#define KBIG (N*HID)   /* 4096 */

__device__ __forceinline__ float silu_f(float x) { return x / (1.0f + expf(-x)); }

// ---------------------------------------------------------------------------
// Geometry: rbf(b,i,j,16), rhat(b,i,j,3), initial f0 (n,64 slot0), f1 (n,16,3 slot0)
// ---------------------------------------------------------------------------
__global__ void __launch_bounds__(N) k_geom(const float* __restrict__ pos,
                                            float* __restrict__ rbf,
                                            float* __restrict__ rhat,
                                            float* __restrict__ f0,
                                            float* __restrict__ f1) {
  int b = blockIdx.x / N, i = blockIdx.x % N, j = threadIdx.x;
  const float* pb = pos + (size_t)b * N * 3;
  float pix = pb[i*3+0], piy = pb[i*3+1], piz = pb[i*3+2];
  float dx = pix - pb[j*3+0], dy = piy - pb[j*3+1], dz = piz - pb[j*3+2];
  float sq = dx*dx + dy*dy + dz*dz;
  float dist = sq > 0.f ? sqrtf(sq) : 0.f;
  float inv = 1.0f / fmaxf(dist, 1e-8f);
  size_t e = (size_t)(b*N + i)*N + j;
  rhat[e*3+0] = dx*inv; rhat[e*3+1] = dy*inv; rhat[e*3+2] = dz*inv;
  float* rb = rbf + e*RBF;
  const float wdt = (5.0f/16.0f)*(5.0f/16.0f);
  #pragma unroll
  for (int c = 0; c < RBF; ++c) {
    float cen = (5.0f/15.0f)*(float)c;
    float t = dist - cen;
    rb[c] = expf(-t*t/wdt);
  }
  if (j == 0) {
    float s = pix*pix + piy*piy + piz*piz;
    f0[(size_t)(b*N+i)*H0] = s > 0.f ? sqrtf(s) : 0.f;
    f1[((size_t)(b*N+i)*H1)*3 + 0] = pix;
    f1[((size_t)(b*N+i)*H1)*3 + 1] = piy;
    f1[((size_t)(b*N+i)*H1)*3 + 2] = piz;
  }
}

__global__ void k_zero(float* s00, float* s11) {
  int t = threadIdx.x;
  if (t < NBATCH*H0) s00[t] = 0.f;
  if (t < NBATCH*48) s11[t] = 0.f;
}

// ---------------------------------------------------------------------------
// Per-edge hidden MLPs (4x 16->32 + silu) + dot[i,j,c] = rhat . f1[j,c,:]
// h00/h11 stored masked + f16 in WMMA-A layout [i][j*32+k]; h10/h01 f32.
// ---------------------------------------------------------------------------
__global__ void __launch_bounds__(N) k_edges(
    int in1,
    const float* __restrict__ w00w, const float* __restrict__ w00b,
    const float* __restrict__ w10w, const float* __restrict__ w10b,
    const float* __restrict__ w01w, const float* __restrict__ w01b,
    const float* __restrict__ w11w, const float* __restrict__ w11b,
    const float* __restrict__ rbf, const float* __restrict__ rhat,
    const float* __restrict__ f1,
    _Float16* __restrict__ h00m, float* __restrict__ h10,
    float* __restrict__ h01, _Float16* __restrict__ h11m,
    float* __restrict__ dotb) {
  __shared__ float sw[4*544];
  const float* srcs[8] = {w00w, w00b, w10w, w10b, w01w, w01b, w11w, w11b};
  for (int m = 0; m < 4; ++m) {
    for (int e = threadIdx.x; e < 512; e += N) sw[m*544 + e] = srcs[2*m][e];
    for (int e = threadIdx.x; e < 32;  e += N) sw[m*544 + 512 + e] = srcs[2*m+1][e];
  }
  __syncthreads();
  int b = blockIdx.x / N, i = blockIdx.x % N, j = threadIdx.x;
  size_t e = (size_t)(b*N + i)*N + j;
  float r[RBF];
  #pragma unroll
  for (int d = 0; d < RBF; ++d) r[d] = rbf[e*RBF + d];
  bool diag = (i == j);
  for (int m = 0; m < 4; ++m) {
    const float* W  = sw + m*544;
    const float* Bv = W + 512;
    for (int k = 0; k < HID; ++k) {
      float a = Bv[k];
      #pragma unroll
      for (int d = 0; d < RBF; ++d) a += r[d] * W[d*HID + k];
      float h = silu_f(a);
      if (m == 0)      h00m[(size_t)(b*N+i)*KBIG + j*HID + k] = (_Float16)(diag ? 0.f : h);
      else if (m == 1) h10[e*HID + k] = h;
      else if (m == 2) h01[e*HID + k] = h;
      else             h11m[(size_t)(b*N+i)*KBIG + j*HID + k] = (_Float16)(diag ? 0.f : h);
    }
  }
  float rh0 = rhat[e*3+0], rh1 = rhat[e*3+1], rh2 = rhat[e*3+2];
  const float* f1j = f1 + (size_t)(b*N + j)*H1*3;
  for (int c = 0; c < in1; ++c)
    dotb[e*16 + c] = rh0*f1j[c*3+0] + rh1*f1j[c*3+1] + rh2*f1j[c*3+2];
}

// ---------------------------------------------------------------------------
// Per-node weight factors: G00[jk,o](f16), G01[j,k,g](f32), G11[jk,gx](f16),
// per-node bias contractions B*, and masked-bias sums S00/S11 via atomics.
// ---------------------------------------------------------------------------
__global__ void __launch_bounds__(128) k_gmat(
    int in0, int in1,
    const float* __restrict__ f0, const float* __restrict__ f1,
    const float* __restrict__ w00l2w, const float* __restrict__ w00l2b,
    const float* __restrict__ w01l2w, const float* __restrict__ w01l2b,
    const float* __restrict__ w11l2w, const float* __restrict__ w11l2b,
    _Float16* __restrict__ G00, float* __restrict__ B00d, float* __restrict__ S00,
    float* __restrict__ G01, float* __restrict__ B01d,
    _Float16* __restrict__ G11, float* __restrict__ B11d, float* __restrict__ S11) {
  __shared__ float sf0[H0];
  __shared__ float sf1[H1*3];
  int b = blockIdx.x / N, j = blockIdx.x % N, t = threadIdx.x;
  if (t < in0)   sf0[t] = f0[(size_t)(b*N+j)*H0 + t];
  if (t < in1*3) sf1[t] = f1[(size_t)(b*N+j)*H1*3 + t];
  __syncthreads();
  for (int e2 = t; e2 < HID*H0; e2 += 128) {
    int k = e2 >> 6, o = e2 & 63;
    float a = 0.f;
    for (int d = 0; d < in0; ++d) a += sf0[d] * w00l2w[(size_t)k*(in0*H0) + d*H0 + o];
    G00[(size_t)b*KBIG*H0 + (size_t)j*HID*H0 + e2] = (_Float16)a;
  }
  if (t < H0) {
    float a = 0.f;
    for (int d = 0; d < in0; ++d) a += sf0[d] * w00l2b[d*H0 + t];
    B00d[(size_t)(b*N+j)*H0 + t] = a;
    atomicAdd(&S00[b*H0 + t], a);
  }
  for (int e2 = t; e2 < HID*H1; e2 += 128) {
    int k = e2 >> 4, g = e2 & 15;
    float a = 0.f;
    for (int d = 0; d < in0; ++d) a += sf0[d] * w01l2w[(size_t)k*(in0*H1) + d*H1 + g];
    G01[(size_t)(b*N+j)*HID*H1 + e2] = a;
  }
  if (t < H1) {
    float a = 0.f;
    for (int d = 0; d < in0; ++d) a += sf0[d] * w01l2b[d*H1 + t];
    B01d[(size_t)(b*N+j)*H1 + t] = a;
  }
  for (int e2 = t; e2 < HID*48; e2 += 128) {
    int k = e2 / 48, gx = e2 % 48, g = gx / 3, x = gx % 3;
    float a = 0.f;
    for (int c = 0; c < in1; ++c) a += sf1[c*3 + x] * w11l2w[(size_t)k*(in1*H1) + c*H1 + g];
    G11[(size_t)b*KBIG*48 + (size_t)j*HID*48 + e2] = (_Float16)a;
  }
  if (t < 48) {
    int g = t / 3, x = t % 3;
    float a = 0.f;
    for (int c = 0; c < in1; ++c) a += sf1[c*3 + x] * w11l2b[c*H1 + g];
    B11d[(size_t)(b*N+j)*48 + t] = a;
    atomicAdd(&S11[b*48 + t], a);
  }
}

// ---------------------------------------------------------------------------
// Big K=4096 GEMM on the WMMA pipe: C[128,ncols] = A[128,4096] x B[4096,ncols]
// One wave per 16x16 tile, 128 chained v_wmma_f32_16x16x32_f16.
// A frag (ISA 16-bit A 16x32): lane<16 -> M=lane, K {0-7,16-23}; lane>=16 -> K {8-15,24-31}
// B frag: lane holds column N=lane%16, K in same order as A frag (strided rows).
// ---------------------------------------------------------------------------
__global__ void __launch_bounds__(32) k_wmma(
    const _Float16* __restrict__ A, const _Float16* __restrict__ Bm,
    float* __restrict__ C, int ncols) {
  int nt = ncols >> 4;
  int ot = blockIdx.x % nt;
  int rest = blockIdx.x / nt;
  int it = rest & 7;
  int b  = rest >> 3;
  int lane = threadIdx.x & 31;
  int m = lane & 15;
  int ksel = lane >> 4;
  const _Float16* ap = A  + (size_t)b*N*KBIG + (size_t)(it*16 + m)*KBIG + ksel*8;
  const _Float16* bp = Bm + (size_t)b*KBIG*ncols + (size_t)(ksel*8)*ncols + ot*16 + m;
  v8f acc = {0.f,0.f,0.f,0.f,0.f,0.f,0.f,0.f};
  for (int kk = 0; kk < KBIG; kk += 32) {
    union { v16h v; float4 q[2]; } ua, ub;
    ua.q[0] = *(const float4*)(ap);
    ua.q[1] = *(const float4*)(ap + 16);
    for (int s = 0; s < 8; ++s) {
      ub.v[s]     = bp[(size_t)s*ncols];
      ub.v[8 + s] = bp[(size_t)(16 + s)*ncols];
    }
    __builtin_prefetch(ap + 128, 0, 1);
    __builtin_prefetch(bp + (size_t)128*ncols, 0, 1);
    acc = __builtin_amdgcn_wmma_f32_16x16x32_f16(false, ua.v, false, ub.v,
                                                 (short)0, acc, false, false);
    ap += 32;
    bp += (size_t)32*ncols;
  }
  float* Cb = C + (size_t)(b*N + it*16)*ncols + ot*16;
  int n  = lane & 15;
  int mo = (lane < 16) ? 0 : 8;
  #pragma unroll
  for (int r = 0; r < 8; ++r) Cb[(mo + r)*ncols + n] = acc[r];
}

// ---------------------------------------------------------------------------
// Per-node i: D[i,c]=sum_j dot;  msg01[i,g,x] via P01 + rhat (LDS atomics);
// U[i,k,c] = sum_j h10*dot  (mask free: diagonal dot/rhat are zero)
// ---------------------------------------------------------------------------
__global__ void __launch_bounds__(N) k_udot(
    int in1,
    const float* __restrict__ h10, const float* __restrict__ h01,
    const float* __restrict__ dotb, const float* __restrict__ G01,
    const float* __restrict__ B01d, const float* __restrict__ rhat,
    float* __restrict__ U, float* __restrict__ Dd, float* __restrict__ M01) {
  __shared__ float smsg[48];
  int b = blockIdx.x / N, i = blockIdx.x % N, t = threadIdx.x;
  if (t < 48) smsg[t] = 0.f;
  __syncthreads();
  size_t e0 = (size_t)(b*N + i)*N;
  if (t < in1) {
    float a = 0.f;
    for (int j = 0; j < N; ++j) a += dotb[(e0 + j)*16 + t];
    Dd[(size_t)(b*N+i)*16 + t] = a;
  }
  {
    int j = t;
    size_t e = e0 + j;
    float hr[HID];
    #pragma unroll
    for (int k = 0; k < HID; ++k) hr[k] = h01[e*HID + k];
    float rh0 = rhat[e*3+0], rh1 = rhat[e*3+1], rh2 = rhat[e*3+2];
    const float* Gj = G01 + (size_t)(b*N + j)*HID*H1;
    const float* Bj = B01d + (size_t)(b*N + j)*H1;
    for (int g = 0; g < H1; ++g) {
      float p = Bj[g];
      #pragma unroll 8
      for (int k = 0; k < HID; ++k) p += hr[k]*Gj[k*H1 + g];
      atomicAdd(&smsg[g*3+0], p*rh0);
      atomicAdd(&smsg[g*3+1], p*rh1);
      atomicAdd(&smsg[g*3+2], p*rh2);
    }
  }
  int total = HID * in1;
  for (int e2 = t; e2 < total; e2 += N) {
    int k = e2 / in1, c = e2 % in1;
    float a = 0.f;
    for (int j = 0; j < N; ++j) a += h10[(e0 + j)*HID + k] * dotb[(e0 + j)*16 + c];
    U[(size_t)(b*N+i)*HID*16 + k*16 + c] = a;
  }
  __syncthreads();
  if (t < 48) M01[(size_t)(b*N+i)*48 + t] = smsg[t];
}

// ---------------------------------------------------------------------------
// f0 epilogue: msg00 (WMMA C + masked bias) + msg10 (U x W2 + D x b2), LayerNorm
// ---------------------------------------------------------------------------
__global__ void __launch_bounds__(H0) k_f0ep(
    int in1, const float* __restrict__ C0, const float* __restrict__ S00,
    const float* __restrict__ B00d, const float* __restrict__ U,
    const float* __restrict__ Dd, const float* __restrict__ w10l2w,
    const float* __restrict__ w10l2b, const float* __restrict__ gam,
    const float* __restrict__ bet, float* __restrict__ f0) {
  __shared__ float red[H0];
  int b = blockIdx.x / N, i = blockIdx.x % N, o = threadIdx.x;
  size_t ib = (size_t)(b*N + i);
  float v = C0[ib*H0 + o] + S00[b*H0 + o] - B00d[ib*H0 + o];
  const float* Uu = U + ib*HID*16;
  for (int k = 0; k < HID; ++k)
    for (int c = 0; c < in1; ++c)
      v += Uu[k*16 + c] * w10l2w[(size_t)k*(in1*H0) + c*H0 + o];
  for (int c = 0; c < in1; ++c) v += Dd[ib*16 + c] * w10l2b[c*H0 + o];
  red[o] = v; __syncthreads();
  for (int s = 32; s > 0; s >>= 1) { if (o < s) red[o] += red[o + s]; __syncthreads(); }
  float mean = red[0] * (1.0f/H0);
  __syncthreads();
  float d = v - mean; red[o] = d*d; __syncthreads();
  for (int s = 32; s > 0; s >>= 1) { if (o < s) red[o] += red[o + s]; __syncthreads(); }
  float var = red[0] * (1.0f/H0);
  f0[ib*H0 + o] = (v - mean) * rsqrtf(var + 1e-5f) * gam[o] + bet[o];
}

// ---------------------------------------------------------------------------
// f1 epilogue: raw = WMMA C1 + masked bias + msg01; gated LayerNorm over norms
// ---------------------------------------------------------------------------
__global__ void __launch_bounds__(48) k_f1ep(
    const float* __restrict__ C1, const float* __restrict__ S11,
    const float* __restrict__ B11d, const float* __restrict__ M01,
    const float* __restrict__ gam, const float* __restrict__ bet,
    float* __restrict__ f1) {
  __shared__ float raw[48];
  __shared__ float nrm[16];
  __shared__ float stat[2];
  int b = blockIdx.x / N, i = blockIdx.x % N, t = threadIdx.x;
  size_t ib = (size_t)(b*N + i);
  float rv = C1[ib*48 + t] + S11[b*48 + t] - B11d[ib*48 + t] + M01[ib*48 + t];
  raw[t] = rv; __syncthreads();
  if (t < 16) {
    float sq = raw[t*3]*raw[t*3] + raw[t*3+1]*raw[t*3+1] + raw[t*3+2]*raw[t*3+2];
    float safe = sq > 0.f ? sqrtf(sq) : 0.f;
    nrm[t] = fmaxf(safe, 1e-8f);
  }
  __syncthreads();
  if (t == 0) {
    float s = 0.f;
    for (int g = 0; g < 16; ++g) s += nrm[g];
    float m = s * (1.0f/16.0f);
    float ss = 0.f;
    for (int g = 0; g < 16; ++g) { float d = nrm[g]-m; ss += d*d; }
    stat[0] = m; stat[1] = rsqrtf(ss*(1.0f/16.0f) + 1e-5f);
  }
  __syncthreads();
  int g = t / 3;
  float scale = (nrm[g] - stat[0]) * stat[1] * gam[g] + bet[g];
  f1[(ib*H1 + g)*3 + (t % 3)] = raw[t] / nrm[g] * scale;
}

// ---------------------------------------------------------------------------
// Max-pool node invariants + 80->128->64->10 classifier (one block per batch)
// ---------------------------------------------------------------------------
__global__ void __launch_bounds__(128) k_head(
    const float* __restrict__ f0, const float* __restrict__ f1,
    const float* __restrict__ rw0, const float* __restrict__ rb0,
    const float* __restrict__ rw1, const float* __restrict__ rb1,
    const float* __restrict__ rw2, const float* __restrict__ rb2,
    float* __restrict__ out) {
  __shared__ float feats[80];
  __shared__ float h1s[128];
  __shared__ float h2s[64];
  int b = blockIdx.x, t = threadIdx.x;
  if (t < 64) {
    float m = -3.4e38f;
    for (int i = 0; i < N; ++i) m = fmaxf(m, f0[(size_t)(b*N+i)*H0 + t]);
    feats[t] = m;
  } else if (t < 80) {
    int g = t - 64;
    float m = -3.4e38f;
    for (int i = 0; i < N; ++i) {
      const float* p = f1 + ((size_t)(b*N+i)*H1 + g)*3;
      float sq = p[0]*p[0] + p[1]*p[1] + p[2]*p[2];
      m = fmaxf(m, sq > 0.f ? sqrtf(sq) : 0.f);
    }
    feats[t] = m;
  }
  __syncthreads();
  {
    float a = rb0[t];
    for (int d = 0; d < 80; ++d) a += feats[d] * rw0[d*128 + t];
    h1s[t] = fmaxf(a, 0.f);
  }
  __syncthreads();
  if (t < 64) {
    float a = rb1[t];
    for (int d = 0; d < 128; ++d) a += h1s[d] * rw1[d*64 + t];
    h2s[t] = fmaxf(a, 0.f);
  }
  __syncthreads();
  if (t < 10) {
    float a = rb2[t];
    for (int d = 0; d < 64; ++d) a += h2s[d] * rw2[d*10 + t];
    out[b*10 + t] = a;
  }
}

// ---------------------------------------------------------------------------
extern "C" void kernel_launch(void* const* d_in, const int* in_sizes, int n_in,
                              void* d_out, int out_size, void* d_ws, size_t ws_size,
                              hipStream_t stream) {
  (void)in_sizes; (void)n_in; (void)out_size; (void)ws_size;
  const float* pos = (const float*)d_in[0];

  char* w = (char*)d_ws;
  auto carve = [&](size_t bytes) -> char* {
    char* p = w; w += (bytes + 255) & ~(size_t)255; return p;
  };
  float*    rbf  = (float*)   carve((size_t)NBATCH*N*N*RBF*4);
  float*    rhat = (float*)   carve((size_t)NBATCH*N*N*3*4);
  float*    f0   = (float*)   carve((size_t)NBATCH*N*H0*4);
  float*    f1   = (float*)   carve((size_t)NBATCH*N*H1*3*4);
  _Float16* h00m = (_Float16*)carve((size_t)NBATCH*N*KBIG*2);
  _Float16* h11m = (_Float16*)carve((size_t)NBATCH*N*KBIG*2);
  float*    h10  = (float*)   carve((size_t)NBATCH*N*N*HID*4);
  float*    h01  = (float*)   carve((size_t)NBATCH*N*N*HID*4);
  float*    dotb = (float*)   carve((size_t)NBATCH*N*N*16*4);
  _Float16* G00  = (_Float16*)carve((size_t)NBATCH*KBIG*H0*2);
  _Float16* G11  = (_Float16*)carve((size_t)NBATCH*KBIG*48*2);
  float*    G01  = (float*)   carve((size_t)NBATCH*N*HID*H1*4);
  float*    B00d = (float*)   carve((size_t)NBATCH*N*H0*4);
  float*    B01d = (float*)   carve((size_t)NBATCH*N*H1*4);
  float*    B11d = (float*)   carve((size_t)NBATCH*N*48*4);
  float*    S00  = (float*)   carve((size_t)NBATCH*H0*4);
  float*    S11  = (float*)   carve((size_t)NBATCH*48*4);
  float*    U    = (float*)   carve((size_t)NBATCH*N*HID*16*4);
  float*    Dd   = (float*)   carve((size_t)NBATCH*N*16*4);
  float*    C0   = (float*)   carve((size_t)NBATCH*N*H0*4);
  float*    C1   = (float*)   carve((size_t)NBATCH*N*48*4);
  float*    M01  = (float*)   carve((size_t)NBATCH*N*48*4);

  k_geom<<<NBATCH*N, N, 0, stream>>>(pos, rbf, rhat, f0, f1);

  int in0 = 1, in1 = 1;
  for (int l = 0; l < 3; ++l) {
    auto P = [&](int k) { return (const float*)d_in[1 + 20*l + k]; };
    k_zero<<<1, 128, 0, stream>>>(S00, S11);
    k_edges<<<NBATCH*N, N, 0, stream>>>(in1, P(0), P(1), P(4), P(5), P(8), P(9),
                                        P(12), P(13), rbf, rhat, f1,
                                        h00m, h10, h01, h11m, dotb);
    k_gmat<<<NBATCH*N, 128, 0, stream>>>(in0, in1, f0, f1, P(2), P(3),
                                         P(10), P(11), P(14), P(15),
                                         G00, B00d, S00, G01, B01d, G11, B11d, S11);
    k_wmma<<<NBATCH*8*(H0/16), 32, 0, stream>>>(h00m, G00, C0, H0);
    k_wmma<<<NBATCH*8*(48/16),  32, 0, stream>>>(h11m, G11, C1, 48);
    k_udot<<<NBATCH*N, N, 0, stream>>>(in1, h10, h01, dotb, G01, B01d, rhat,
                                       U, Dd, M01);
    k_f0ep<<<NBATCH*N, H0, 0, stream>>>(in1, C0, S00, B00d, U, Dd,
                                        P(6), P(7), P(16), P(17), f0);
    k_f1ep<<<NBATCH*N, 48, 0, stream>>>(C1, S11, B11d, M01, P(18), P(19), f1);
    in0 = H0; in1 = H1;
  }

  k_head<<<NBATCH, 128, 0, stream>>>(f0, f1,
      (const float*)d_in[61], (const float*)d_in[62],
      (const float*)d_in[63], (const float*)d_in[64],
      (const float*)d_in[65], (const float*)d_in[66],
      (float*)d_out);
}
